// EncoderLayer_71159018160993
// MI455X (gfx1250) — matrix operand order repaired and checked by reading, not compiled, not run on hardware
//
#include <hip/hip_runtime.h>

// ---------------------------------------------------------------------------
// Types for CDNA5 WMMA / TDM (wave32)
// ---------------------------------------------------------------------------
typedef __attribute__((ext_vector_type(16))) _Float16 v16h;
typedef __attribute__((ext_vector_type(8)))  _Float16 v8h;
typedef __attribute__((ext_vector_type(4)))  _Float16 v4h;
typedef __attribute__((ext_vector_type(8)))  float    v8f;
typedef __attribute__((ext_vector_type(4)))  unsigned int u32x4;
typedef __attribute__((ext_vector_type(4)))  int  i32x4;
typedef __attribute__((ext_vector_type(8)))  int  i32x8;

union HFrag { v16h v; _Float16 h[16]; };
union FFrag { v8f  v; float    f[8];  };

__device__ __forceinline__ v8f wmma_f16(v16h a, v16h b, v8f c) {
    return __builtin_amdgcn_wmma_f32_16x16x32_f16(
        false, a, false, b, (short)0, c, false, false);
}

__device__ __forceinline__ void load8(HFrag& d, int base, const _Float16* p) {
    const v8h x = *reinterpret_cast<const v8h*>(p);
#pragma unroll
    for (int i = 0; i < 8; ++i) d.h[base + i] = x[i];
}

// Low 32 bits of a flat pointer into LDS == LDS byte offset (aperture layout).
__device__ __forceinline__ unsigned lds_off_of(const void* p) {
    return (unsigned)(uintptr_t)p;
}

// ---------------------------------------------------------------------------
// Tensor Data Mover: DMA a 2D f16 tile (tile_d0 contiguous elems x tile_d1
// rows, row stride d0_stride elems) from global into LDS. Wave-scalar op,
// tracked by TENSORcnt.
// ---------------------------------------------------------------------------
__device__ __forceinline__ void tdm_load_2d_f16(unsigned lds_off, const _Float16* g,
                                                unsigned tile_d0, unsigned tile_d1,
                                                unsigned tens_d0, unsigned tens_d1,
                                                unsigned d0_stride) {
    const unsigned long long ga = (unsigned long long)(uintptr_t)g;
    u32x4 g0;
    g0[0] = 1u;                                   // count=1, user descriptor
    g0[1] = lds_off;                              // lds_addr (bytes)
    g0[2] = (unsigned)(ga & 0xFFFFFFFFu);         // global_addr[31:0]
    g0[3] = (unsigned)((ga >> 32) & 0x1FFFFFFu)   // global_addr[56:32]
          | (2u << 30);                           // type = 2 ("image")
    i32x8 g1;
    g1[0] = (int)(1u << 16);                      // data_size=1 (2 bytes)
    g1[1] = (int)((tens_d0 & 0xFFFFu) << 16);                         // dim0 lo
    g1[2] = (int)(((tens_d0 >> 16) & 0xFFFFu) | ((tens_d1 & 0xFFFFu) << 16));
    g1[3] = (int)(((tens_d1 >> 16) & 0xFFFFu) | ((tile_d0 & 0xFFFFu) << 16));
    g1[4] = (int)(tile_d1 & 0xFFFFu);             // tile_dim1 (tile_dim2=0)
    g1[5] = (int)d0_stride;                       // tensor_dim0_stride[31:0]
    g1[6] = 0;
    g1[7] = 0;
    i32x4 z4 = {0, 0, 0, 0};
#if __clang_major__ >= 23
    i32x8 z8 = {0, 0, 0, 0, 0, 0, 0, 0};
    __builtin_amdgcn_tensor_load_to_lds(g0, g1, z4, z4, z8, 0);
#else
    __builtin_amdgcn_tensor_load_to_lds(g0, g1, z4, z4, 0);
#endif
}

// ---------------------------------------------------------------------------
// f32 -> f16 conversion (vectorized, n multiple of 4)
// ---------------------------------------------------------------------------
__global__ void cvt_f32_f16(const float* __restrict__ in, _Float16* __restrict__ out,
                            int n4) {
    const int i = blockIdx.x * blockDim.x + threadIdx.x;
    if (i >= n4) return;
    const float4 x = reinterpret_cast<const float4*>(in)[i];
    v4h o;
    o[0] = (_Float16)x.x; o[1] = (_Float16)x.y;
    o[2] = (_Float16)x.z; o[3] = (_Float16)x.w;
    reinterpret_cast<v4h*>(out)[i] = o;
}

// ---------------------------------------------------------------------------
// Block-tiled WMMA GEMM with TDM double buffering.
//   Out[m,n] = act( sum_k A[m,k] * W[n,k] + bias[n] )
// A: MxK f16, W: NxK f16 (both row-major). Block = 256 thr (8 waves) computes
// a 128x128 tile; each wave a 64x32 macro-tile (8 WMMAs / K-step). K step 32.
// ---------------------------------------------------------------------------
template <typename OT, bool RELU>
__global__ __launch_bounds__(256) void wmma_gemm_tdm(
    const _Float16* __restrict__ A, const _Float16* __restrict__ W,
    const float* __restrict__ bias, OT* __restrict__ Out, int M, int N, int K) {
    __shared__ _Float16 smA[2][128 * 32];
    __shared__ _Float16 smB[2][128 * 32];

    const int tilesN = N >> 7;
    const int tm = blockIdx.x / tilesN;
    const int tn = blockIdx.x % tilesN;
    const int wid  = threadIdx.x >> 5;
    const int lane = threadIdx.x & 31;
    const int half = lane >> 4;
    const int r    = lane & 15;
    const int wm = (wid >> 2) << 6;   // 0 / 64
    const int wn = (wid & 3) << 5;    // 0 / 32 / 64 / 96

    const _Float16* Ab = A + (size_t)tm * 128 * K;
    const _Float16* Wb = W + (size_t)tn * 128 * K;

    if (wid == 0) {  // TDM: stage 0, A and B tiles (128 rows x 32 k)
        tdm_load_2d_f16(lds_off_of(&smA[0][0]), Ab, 32, 128, (unsigned)K, 4096u, (unsigned)K);
        tdm_load_2d_f16(lds_off_of(&smB[0][0]), Wb, 32, 128, (unsigned)K, 4096u, (unsigned)K);
    }

    FFrag acc[4][2] = {};
    const int nk = K >> 5;
    for (int s = 0; s < nk; ++s) {
        const int buf = s & 1;
        if (wid == 0) {
            if (s + 1 < nk) {  // prefetch next stage, then wait for current
                tdm_load_2d_f16(lds_off_of(&smA[buf ^ 1][0]), Ab + (size_t)(s + 1) * 32,
                                32, 128, (unsigned)K, 4096u, (unsigned)K);
                tdm_load_2d_f16(lds_off_of(&smB[buf ^ 1][0]), Wb + (size_t)(s + 1) * 32,
                                32, 128, (unsigned)K, 4096u, (unsigned)K);
                __builtin_amdgcn_s_wait_tensorcnt(2);
            } else {
                __builtin_amdgcn_s_wait_tensorcnt(0);
            }
        }
        __syncthreads();

        const _Float16* As = &smA[buf][0];
        const _Float16* Bs = &smB[buf][0];
        HFrag af[4], bf[2];
#pragma unroll
        for (int ms = 0; ms < 4; ++ms) {
            const _Float16* p = As + (wm + ms * 16 + r) * 32;
            load8(af[ms], 0, p + half * 8);
            load8(af[ms], 8, p + 16 + half * 8);
        }
#pragma unroll
        for (int ns = 0; ns < 2; ++ns) {
            const _Float16* p = Bs + (wn + ns * 16 + r) * 32 + half * 8;
            load8(bf[ns], 0, p);
            load8(bf[ns], 8, p + 16);
        }
#pragma unroll
        for (int ms = 0; ms < 4; ++ms)
#pragma unroll
            for (int ns = 0; ns < 2; ++ns)
                acc[ms][ns].v = wmma_f16(af[ms].v, bf[ns].v, acc[ms][ns].v);
        __syncthreads();
    }

#pragma unroll
    for (int ns = 0; ns < 2; ++ns) {
        const int n  = tn * 128 + wn + ns * 16 + r;
        const float bv = bias[n];
#pragma unroll
        for (int ms = 0; ms < 4; ++ms) {
#pragma unroll
            for (int j = 0; j < 8; ++j) {
                float val = acc[ms][ns].f[j] + bv;
                if (RELU) val = fmaxf(val, 0.0f);
                Out[(size_t)(tm * 128 + wm + ms * 16 + half * 8 + j) * N + n] = (OT)val;
            }
        }
    }
}

// ---------------------------------------------------------------------------
// Fused attention, one block (4 waves) per (head, 16-query rows).
// Phase 1: scores into LDS via WMMA. Phase 2: f32 softmax, coalesced
// non-temporal write of the attn tensor. Phase 3: context = P*V with the
// 32x64 V tile TDM-double-buffered through LDS.
// ---------------------------------------------------------------------------
#define ATTN_SLD 2064  // padded LDS row stride (floats); 2064 % 64 = 16

__global__ __launch_bounds__(128) void attn_kernel(
    const _Float16* __restrict__ qh, const _Float16* __restrict__ kh,
    const _Float16* __restrict__ vh, float* __restrict__ attn_out,
    float* __restrict__ ctx) {
    extern __shared__ float sc[];                    // [16][ATTN_SLD]
    _Float16* vbuf = reinterpret_cast<_Float16*>(sc + 16 * ATTN_SLD);  // 2*[32*64]
    __shared__ float red[16][9];
    __shared__ float rowstat[16];

    const int tid  = threadIdx.x;
    const int w4   = tid >> 5;
    const int lane = tid & 31;
    const int half = lane >> 4;
    const int r    = lane & 15;

    const int bh = blockIdx.y;
    const int b  = bh >> 4;
    const int h  = bh & 15;
    const int q0 = blockIdx.x * 16;
    const size_t hoff = (size_t)b * 2048 * 1024 + (size_t)h * 64;

    // --- preload Q fragments (16 rows x 64, two K=32 chunks) ---
    HFrag a0, a1;
    {
        const _Float16* qp = qh + hoff + (size_t)(q0 + r) * 1024;
        load8(a0, 0, qp + half * 8);
        load8(a0, 8, qp + 16 + half * 8);
        load8(a1, 0, qp + 32 + half * 8);
        load8(a1, 8, qp + 48 + half * 8);
    }

    // --- phase 1: scores = (Q K^T) * D^-0.5 into LDS ---
    const float scale = 0.03125f;  // 1024^-0.5 (reference scales by model dim)
    for (int t = w4; t < 128; t += 4) {
        const _Float16* kp = kh + hoff + (size_t)(t * 16 + r) * 1024 + half * 8;
        HFrag b0, b1;
        load8(b0, 0, kp);      load8(b0, 8, kp + 16);
        load8(b1, 0, kp + 32); load8(b1, 8, kp + 48);
        FFrag c = {};
        c.v = wmma_f16(a0.v, b0.v, c.v);
        c.v = wmma_f16(a1.v, b1.v, c.v);
#pragma unroll
        for (int j = 0; j < 8; ++j)
            sc[(half * 8 + j) * ATTN_SLD + t * 16 + r] = c.f[j] * scale;
    }
    __syncthreads();

    // --- phase 2: row softmax over 2048 cols ---
    {
        const int row = tid >> 3;
        const int sub = tid & 7;
        float* srow = sc + row * ATTN_SLD;

        float m = -3.4e38f;
        for (int c = sub; c < 2048; c += 8) m = fmaxf(m, srow[c]);
        red[row][sub] = m;
        __syncthreads();
        if (sub == 0) {
            float mm = red[row][0];
#pragma unroll
            for (int i = 1; i < 8; ++i) mm = fmaxf(mm, red[row][i]);
            rowstat[row] = mm;
        }
        __syncthreads();
        const float mx = rowstat[row];

        float s = 0.0f;
        for (int c = sub; c < 2048; c += 8) {
            float e = __expf(srow[c] - mx);
            srow[c] = e;
            s += e;
        }
        red[row][sub] = s;
        __syncthreads();
        if (sub == 0) {
            float ss = 0.0f;
#pragma unroll
            for (int i = 0; i < 8; ++i) ss += red[row][i];
            rowstat[row] = 1.0f / ss;
        }
    }
    __syncthreads();

    // coalesced, non-temporal write of attn rows (write-once 537MB tensor)
    for (int row = 0; row < 16; ++row) {
        const float inv = rowstat[row];
        float* srow = sc + row * ATTN_SLD;
        float* arow = attn_out + ((size_t)bh * 2048 + q0 + row) * 2048;
        for (int c = tid; c < 2048; c += 128) {
            const float p = srow[c] * inv;
            srow[c] = p;
            __builtin_nontemporal_store(p, arow + c);
        }
    }
    __syncthreads();

    // --- phase 3: context = P (16x2048) * V (2048x64), V staged via TDM ---
    if (w4 == 0)  // stage 0: V rows 0..31, all 64 head cols (row stride 1024)
        tdm_load_2d_f16(lds_off_of(vbuf), vh + hoff, 64, 32, 1024u, 2048u, 1024u);

    const int n0 = w4 * 16;
    FFrag acc = {};
    for (int kb = 0; kb < 2048; kb += 32) {
        const int buf = (kb >> 5) & 1;
        if (w4 == 0) {
            if (kb + 32 < 2048) {
                tdm_load_2d_f16(lds_off_of(vbuf + (buf ^ 1) * (32 * 64)),
                                vh + hoff + (size_t)(kb + 32) * 1024,
                                64, 32, 1024u, 2048u, 1024u);
                __builtin_amdgcn_s_wait_tensorcnt(1);
            } else {
                __builtin_amdgcn_s_wait_tensorcnt(0);
            }
        }
        __syncthreads();

        HFrag a, bf;
#pragma unroll
        for (int i = 0; i < 8; ++i) {
            a.h[i]     = (_Float16)sc[r * ATTN_SLD + kb + half * 8 + i];
            a.h[8 + i] = (_Float16)sc[r * ATTN_SLD + kb + 16 + half * 8 + i];
        }
        const _Float16* vb = vbuf + buf * (32 * 64);
#pragma unroll
        for (int i = 0; i < 16; ++i) bf.h[i] = vb[(half * 8 + (i & 7) + ((i >> 3) << 4)) * 64 + n0 + r];
        acc.v = wmma_f16(a.v, bf.v, acc.v);
        __syncthreads();
    }
#pragma unroll
    for (int j = 0; j < 8; ++j)
        ctx[(size_t)(b * 2048 + q0 + half * 8 + j) * 1024 + h * 64 + n0 + r] = acc.f[j];
}

// ---------------------------------------------------------------------------
// out = LayerNorm(X + Y); optionally also emit f16 copy of out.
// CUSTOM=false: torch LN (biased var, eps=1e-5 inside sqrt).
// CUSTOM=true:  repo LN (ddof=1 std, eps=1e-6 added to std).
// ---------------------------------------------------------------------------
template <bool CUSTOM>
__global__ void add_ln_kernel(const float* __restrict__ X, const float* __restrict__ Y,
                              const float* __restrict__ g, const float* __restrict__ bb,
                              float* __restrict__ out, _Float16* __restrict__ out16,
                              int Dm) {
    __shared__ float red[256];
    const int row = blockIdx.x;
    const int tid = threadIdx.x;
    const float* xr = X + (size_t)row * Dm;
    const float* yr = Y + (size_t)row * Dm;

    float s = 0.0f;
    for (int c = tid; c < Dm; c += 256) s += xr[c] + yr[c];
    red[tid] = s;
    __syncthreads();
    for (int off = 128; off > 0; off >>= 1) {
        if (tid < off) red[tid] += red[tid + off];
        __syncthreads();
    }
    const float mean = red[0] / Dm;
    __syncthreads();

    float sq = 0.0f;
    for (int c = tid; c < Dm; c += 256) {
        float v = xr[c] + yr[c] - mean;
        sq += v * v;
    }
    red[tid] = sq;
    __syncthreads();
    for (int off = 128; off > 0; off >>= 1) {
        if (tid < off) red[tid] += red[tid + off];
        __syncthreads();
    }
    float denom;
    if (CUSTOM) denom = sqrtf(red[0] / (Dm - 1)) + 1e-6f;
    else        denom = sqrtf(red[0] / Dm + 1e-5f);
    const float inv = 1.0f / denom;

    for (int c = tid; c < Dm; c += 256) {
        float v = xr[c] + yr[c];
        float o = g[c] * (v - mean) * inv + bb[c];
        out[(size_t)row * Dm + c] = o;
        if (out16) out16[(size_t)row * Dm + c] = (_Float16)o;
    }
}

// ---------------------------------------------------------------------------
// Host-side orchestration
// ---------------------------------------------------------------------------
extern "C" void kernel_launch(void* const* d_in, const int* in_sizes, int n_in,
                              void* d_out, int out_size, void* d_ws, size_t ws_size,
                              hipStream_t stream) {
    (void)in_sizes; (void)n_in; (void)out_size; (void)ws_size;

    constexpr int cB = 2, cS = 2048, cD = 1024, cH = 16, cF = 4096;
    constexpr int BS = cB * cS;  // 4096 rows

    const float* x     = (const float*)d_in[0];
    const float* wq    = (const float*)d_in[1];
    const float* bq    = (const float*)d_in[2];
    const float* wk    = (const float*)d_in[3];
    const float* bk    = (const float*)d_in[4];
    const float* wv    = (const float*)d_in[5];
    const float* bv    = (const float*)d_in[6];
    const float* ln1_g = (const float*)d_in[7];
    const float* ln1_b = (const float*)d_in[8];
    const float* w1    = (const float*)d_in[9];
    const float* b1    = (const float*)d_in[10];
    const float* w2    = (const float*)d_in[11];
    const float* b2    = (const float*)d_in[12];
    const float* ffn_g = (const float*)d_in[13];
    const float* ffn_b = (const float*)d_in[14];

    float* out      = (float*)d_out;              // [B,S,D]
    float* attn_out = out + (size_t)BS * cD;      // [B*H,S,S]

    // Workspace carve-up (~142 MB)
    _Float16* xh   = (_Float16*)d_ws;
    _Float16* wqh  = xh  + (size_t)BS * cD;
    _Float16* wkh  = wqh + (size_t)cD * cD;
    _Float16* wvh  = wkh + (size_t)cD * cD;
    _Float16* w1h  = wvh + (size_t)cD * cD;
    _Float16* w2h  = w1h + (size_t)cF * cD;
    _Float16* qhp  = w2h + (size_t)cD * cF;
    _Float16* khp  = qhp + (size_t)BS * cD;
    _Float16* vhp  = khp + (size_t)BS * cD;
    _Float16* mhah = vhp + (size_t)BS * cD;
    _Float16* hb   = mhah + (size_t)BS * cD;
    float*    ctx  = (float*)(hb + (size_t)BS * cF);
    float*    mha  = ctx + (size_t)BS * cD;
    float*    yb   = mha + (size_t)BS * cD;

    // 1) convert GEMM operands to f16 once
    auto cvt = [&](const float* src, _Float16* dst, size_t n) {
        const int n4 = (int)(n / 4);
        cvt_f32_f16<<<(n4 + 255) / 256, 256, 0, stream>>>(src, dst, n4);
    };
    cvt(x,  xh,  (size_t)BS * cD);
    cvt(wq, wqh, (size_t)cD * cD);
    cvt(wk, wkh, (size_t)cD * cD);
    cvt(wv, wvh, (size_t)cD * cD);
    cvt(w1, w1h, (size_t)cF * cD);
    cvt(w2, w2h, (size_t)cD * cF);

    // 2) Q/K/V projections (TDM-staged block GEMM, f16 out)
    const dim3 gblk(256);
    const int blocksQKV = (BS / 128) * (cD / 128);   // 256
    wmma_gemm_tdm<_Float16, false><<<blocksQKV, gblk, 0, stream>>>(xh, wqh, bq, qhp, BS, cD, cD);
    wmma_gemm_tdm<_Float16, false><<<blocksQKV, gblk, 0, stream>>>(xh, wkh, bk, khp, BS, cD, cD);
    wmma_gemm_tdm<_Float16, false><<<blocksQKV, gblk, 0, stream>>>(xh, wvh, bv, vhp, BS, cD, cD);

    // 3) fused attention
    const size_t smem = (size_t)16 * ATTN_SLD * sizeof(float)   // scores
                      + (size_t)2 * 32 * 64 * sizeof(_Float16); // V double buffer
    attn_kernel<<<dim3(cS / 16, cB * cH), dim3(128), smem, stream>>>(qhp, khp, vhp,
                                                                     attn_out, ctx);

    // 4) mha_out = torch-LN(x + context), f32 + f16 copies
    add_ln_kernel<false><<<BS, 256, 0, stream>>>(x, ctx, ln1_g, ln1_b, mha, mhah, cD);

    // 5) FFN
    const int blocksF1 = (BS / 128) * (cF / 128);    // 1024
    const int blocksF2 = (BS / 128) * (cD / 128);    // 256
    wmma_gemm_tdm<_Float16, true ><<<blocksF1, gblk, 0, stream>>>(mhah, w1h, b1, hb, BS, cF, cD);
    wmma_gemm_tdm<float,    false><<<blocksF2, gblk, 0, stream>>>(hb, w2h, b2, yb, BS, cD, cF);

    // 6) out = custom-LN(mha_out + y)
    add_ln_kernel<true><<<BS, 256, 0, stream>>>(mha, yb, ffn_g, ffn_b, out, nullptr, cD);
}
